// AdaptiveSparseSelfAttention_53644141527454
// MI455X (gfx1250) — matrix operand order repaired and checked by reading, not compiled, not run on hardware
//
#include <hip/hip_runtime.h>
#include <hip/hip_bf16.h>

// ---------------------------------------------------------------------------
// Adaptive sparse self-attention, CDNA5 (gfx1250), wave32 + WMMA f16->f32.
//   B=4, T=2048, DIM=512, H=8, dk=64, TOPK=64
// Pipeline:
//   1) fp32 -> f16 convert (x) ; fp32 -> f16 convert+transpose (Wqkv, Wout)
//   2) qkv = xh @ Wqkv           (WMMA GEMM, Bt layout, f16 out, 16x64/wave)
//   3) V extracted + transposed to [B,H,dk,T] for vectorized PV loads
//   4) per (b,h,16-query-tile) attention: S panel in 128KB LDS,
//      dense softmax + top-64 threshold (bisection) + gate-combined P,
//      single PV WMMA matmul
//   5) out = attn_h @ Wout       (WMMA GEMM, f32 out)
// All WMMA operand fetches are 128-bit (global_load_b128 / ds_load_b128).
// ---------------------------------------------------------------------------

typedef __attribute__((ext_vector_type(16))) _Float16 v16h;
typedef __attribute__((ext_vector_type(8)))  _Float16 v8h;
typedef __attribute__((ext_vector_type(8)))  float    v8f;
typedef __attribute__((ext_vector_type(4)))  float    v4f;

#define BB   4
#define TT   2048
#define HH   8
#define DKK  64
#define TOPK 64
#define DIMM 512
#define C3   1536   // 3*DIM, row stride of qkv

static __device__ __forceinline__ v8h  ld8h (const _Float16* p) { return *(const v8h*)p; }
static __device__ __forceinline__ v16h ld16h(const _Float16* p) { return *(const v16h*)p; }
static __device__ __forceinline__ v8f  lds8f(const float* p)    { return *(const v8f*)p; }
static __device__ __forceinline__ v4f  lds4f(const float* p)    { return *(const v4f*)p; }

// A-fragment (16x32 f16): per lane two contiguous 8-half runs at +16g and +16g+8
static __device__ __forceinline__ v16h load_afrag(const _Float16* row, int g)
{
    v8h lo = ld8h(row + 16*g);
    v8h hi = ld8h(row + 16*g + 8);
    return __builtin_shufflevector(lo, hi, 0,1,2,3,4,5,6,7,8,9,10,11,12,13,14,15);
}

// ---------------------------------------------------------------- convert ---
__global__ void __launch_bounds__(256)
f32_to_f16_kernel(const float* __restrict__ in, _Float16* __restrict__ out, int n)
{
    int i = blockIdx.x * blockDim.x + threadIdx.x;
    if (i < n) out[i] = (_Float16)in[i];
}

// out_t[n*K + k] = (f16) in[k*N + n]   (weight [K,N] -> [N,K] transposed f16)
__global__ void __launch_bounds__(256)
f32_to_f16_transpose_kernel(const float* __restrict__ in, _Float16* __restrict__ out,
                            int K, int N)
{
    int i = blockIdx.x * blockDim.x + threadIdx.x;
    if (i >= N * K) return;
    int n = i / K, k = i % K;
    out[i] = (_Float16)in[(long)k * N + n];
}

// Vt[((b*H+h)*64 + d)*T + s] = qkv[(b*T+s)*1536 + 1024 + h*64 + d]
__global__ void __launch_bounds__(256)
v_transpose_kernel(const _Float16* __restrict__ qkv, _Float16* __restrict__ vt)
{
    int i = blockIdx.x * blockDim.x + threadIdx.x;   // over B*H*64*T = 4M
    if (i >= BB * HH * DKK * TT) return;
    int s = i & (TT - 1);
    int d = (i >> 11) & 63;
    int h = (i >> 17) & 7;
    int b = i >> 20;
    vt[i] = qkv[(long)(b*TT + s)*C3 + 2*DIMM + h*DKK + d];
}

// ------------------------------------------------------------- WMMA GEMM ---
// C[M,N] = A[M,K] @ Bt[N,K]^T.  One wave -> 16x64 tile (A reuse x4).
template <bool F16OUT>
__global__ void __launch_bounds__(256)
gemm_f16_wmma(const _Float16* __restrict__ A,   // [M,K]
              const _Float16* __restrict__ Bt,  // [N,K] (transposed B)
              void* __restrict__ Cout, int M, int N, int K)
{
    const int wid     = (blockIdx.x * blockDim.x + threadIdx.x) >> 5;
    const int ntilesN = N >> 6;                   // 64-wide N tiles
    const int ntiles  = (M >> 4) * ntilesN;
    if (wid >= ntiles) return;                    // wave-uniform

    const int tm   = (wid / ntilesN) << 4;
    const int tn   = (wid % ntilesN) << 6;
    const int lane = threadIdx.x & 31;
    const int g    = lane >> 4;
    const int mr   = lane & 15;

    const _Float16* arow = A + (long)(tm + mr) * K;

    v8f acc[4] = { {}, {}, {}, {} };
    for (int k0 = 0; k0 < K; k0 += 32) {
        const v16h af = load_afrag(arow + k0, g);
#pragma unroll
        for (int nt = 0; nt < 4; ++nt) {
            // B-fragment: lane col = tn+nt*16+mr, K run 16g..16g+15 contiguous
            const v16h bf = ld16h(Bt + (long)(tn + nt*16 + mr) * K + k0 + 16*g);
            acc[nt] = __builtin_amdgcn_wmma_f32_16x16x32_f16(false, af, false, bf,
                                                             (short)0, acc[nt],
                                                             false, false);
        }
    }

#pragma unroll
    for (int nt = 0; nt < 4; ++nt) {
#pragma unroll
        for (int r = 0; r < 8; ++r) {
            const long m = tm + r + 8*g;
            const long n = tn + nt*16 + mr;
            if constexpr (F16OUT) ((_Float16*)Cout)[m * N + n] = (_Float16)acc[nt][r];
            else                  ((float*)   Cout)[m * N + n] = acc[nt][r];
        }
    }
}

// ------------------------------------------------------------- attention ---
// One block = 4 wave32 = 128 threads handles (b, h, 16 query rows).
// Dynamic LDS: S[16][2048] f32 (128KB) + scratch + out accumulator.
__global__ void __launch_bounds__(128)
attn_topk_kernel(const _Float16* __restrict__ qkv,     // [B*T, 1536]
                 const _Float16* __restrict__ vt,      // [B,H,64,T]
                 const float*   __restrict__ alpha,    // [H]
                 _Float16*      __restrict__ attn_out) // [B*T, 512] f16
{
    extern __shared__ float smem[];
    float* S      = smem;                    // 16*2048
    float* red    = smem + 16*2048;          // 16*8
    float* rowmax = red + 128;               // 16
    float* sumd   = rowmax + 16;             // 16
    float* sums   = sumd + 16;               // 16
    float* tlo    = sums + 16;               // 16
    float* thi    = tlo + 16;                // 16
    float* outacc = thi + 16;                // 16*64

    const int qt = blockIdx.x & 127;         // T/16 query tiles
    const int h  = (blockIdx.x >> 7) & 7;
    const int b  = blockIdx.x >> 10;
    const int t0 = qt << 4;

    const int tid  = threadIdx.x;
    const int wv   = tid >> 5;
    const int lane = tid & 31;
    const int g    = lane >> 4;
    const int mr   = lane & 15;

    // ---- Q fragments (dk 0..31, 32..63) : 4x global_load_b128 per lane ----
    const _Float16* qptr = qkv + (long)(b*TT + t0 + mr) * C3 + h*DKK;
    const v16h a0 = load_afrag(qptr, g);
    const v16h a1 = load_afrag(qptr + 32, g);

    // ---- Phase 1: S[:, wv*512 .. +511] = (Q @ K^T) / 8 ----
    for (int t = 0; t < 32; ++t) {
        const int s0 = wv*512 + t*16;
        const _Float16* kptr = qkv + (long)(b*TT + s0 + mr) * C3 + DIMM + h*DKK;
        if (t + 1 < 32)
            __builtin_prefetch((const void*)(kptr + 16 * C3), 0, 1);
        // B-fragment: lane col = key s0+mr, dk run 16g..16g+15 contiguous
        const v16h b0 = ld16h(kptr + 16*g);
        const v16h b1 = ld16h(kptr + 32 + 16*g);
        v8f c = {};
        c = __builtin_amdgcn_wmma_f32_16x16x32_f16(false, a0, false, b0,
                                                   (short)0, c, false, false);
        c = __builtin_amdgcn_wmma_f32_16x16x32_f16(false, a1, false, b1,
                                                   (short)0, c, false, false);
#pragma unroll
        for (int r = 0; r < 8; ++r)
            S[(r + 8*g)*TT + s0 + mr] = c[r] * 0.125f;   // 1/sqrt(64)
    }
    __syncthreads();

    // ---- Phase 2: row max + dense softmax denominator (v4f LDS scans) ----
    const int r = tid & 15;
    const int c = tid >> 4;                   // 8 chunks of 256 columns
    const float* rowp = S + r*TT + c*256;

    float mx = -1e30f;
    for (int i = 0; i < 256; i += 4) {
        v4f v = lds4f(rowp + i);
        mx = fmaxf(mx, fmaxf(fmaxf(v.x, v.y), fmaxf(v.z, v.w)));
    }
    red[r*8 + c] = mx;
    __syncthreads();
    if (c == 0) {
        float m2 = red[r*8];
        for (int i = 1; i < 8; ++i) m2 = fmaxf(m2, red[r*8 + i]);
        rowmax[r] = m2;
    }
    __syncthreads();
    const float rm = rowmax[r];

    float sm = 0.f;
    for (int i = 0; i < 256; i += 4) {
        v4f v = lds4f(rowp + i);
        sm += __expf(v.x - rm) + __expf(v.y - rm) + __expf(v.z - rm) + __expf(v.w - rm);
    }
    red[r*8 + c] = sm;
    __syncthreads();
    if (c == 0) {
        float s2 = 0.f;
        for (int i = 0; i < 8; ++i) s2 += red[r*8 + i];
        sumd[r] = s2;
        tlo[r]  = rm - 88.f;
        thi[r]  = rm;
    }
    __syncthreads();

    // ---- Phase 3: top-64 threshold via 24-step value bisection ----
    for (int it = 0; it < 24; ++it) {
        const float mid = 0.5f * (tlo[r] + thi[r]);
        float cnt = 0.f;
        for (int i = 0; i < 256; i += 4) {
            v4f v = lds4f(rowp + i);
            cnt += (v.x >= mid) + (v.y >= mid) + (v.z >= mid) + (v.w >= mid);
        }
        red[r*8 + c] = cnt;
        __syncthreads();
        if (c == 0) {
            float tot = 0.f;
            for (int i = 0; i < 8; ++i) tot += red[r*8 + i];
            if (tot >= (float)TOPK) tlo[r] = mid; else thi[r] = mid;
        }
        __syncthreads();
    }
    const float thr = tlo[r];

    // ---- Phase 4: sparse softmax denominator ----
    float ssum = 0.f;
    for (int i = 0; i < 256; i += 4) {
        v4f v = lds4f(rowp + i);
        ssum += (v.x >= thr) ? __expf(v.x - rm) : 0.f;
        ssum += (v.y >= thr) ? __expf(v.y - rm) : 0.f;
        ssum += (v.z >= thr) ? __expf(v.z - rm) : 0.f;
        ssum += (v.w >= thr) ? __expf(v.w - rm) : 0.f;
    }
    red[r*8 + c] = ssum;
    __syncthreads();
    if (c == 0) {
        float s2 = 0.f;
        for (int i = 0; i < 8; ++i) s2 += red[r*8 + i];
        sums[r] = fmaxf(s2, 1e-30f);
    }
    __syncthreads();

    // ---- Phase 5: overwrite S with gate-combined probabilities ----
    const float gate  = 1.f / (1.f + __expf(-alpha[h]));
    const float inv_d = 1.f / sumd[r];
    const float inv_s = 1.f / sums[r];
    float* roww = S + r*TT + c*256;
    for (int i = 0; i < 256; i += 4) {
        v4f v = lds4f(roww + i);
        v4f o;
        o.x = gate * __expf(v.x - rm) * inv_d + ((v.x >= thr) ? (1.f-gate) * __expf(v.x - rm) * inv_s : 0.f);
        o.y = gate * __expf(v.y - rm) * inv_d + ((v.y >= thr) ? (1.f-gate) * __expf(v.y - rm) * inv_s : 0.f);
        o.z = gate * __expf(v.z - rm) * inv_d + ((v.z >= thr) ? (1.f-gate) * __expf(v.z - rm) * inv_s : 0.f);
        o.w = gate * __expf(v.w - rm) * inv_d + ((v.w >= thr) ? (1.f-gate) * __expf(v.w - rm) * inv_s : 0.f);
        *(v4f*)(roww + i) = o;
    }
    for (int i = tid; i < 16*64; i += 128) outacc[i] = 0.f;
    __syncthreads();

    // ---- Phase 6: D[16x64] = P[16x2048] @ V[2048x64], K split over 4 waves
    const _Float16* vhead = vt + (long)((b*HH + h) * DKK) * TT;
    v8f acc[4] = { {}, {}, {}, {} };
    for (int ks = 0; ks < 16; ++ks) {
        const int kb = wv*512 + ks*32;
        // A-fragment from LDS: two contiguous 8-float runs -> ds_load_b128 x4
        const v8f plo = lds8f(S + mr*TT + kb + 16*g);
        const v8f phi = lds8f(S + mr*TT + kb + 16*g + 8);
        v16h pa;
#pragma unroll
        for (int e = 0; e < 8; ++e) {
            pa[e]     = (_Float16)plo[e];
            pa[e + 8] = (_Float16)phi[e];
        }
#pragma unroll
        for (int nt = 0; nt < 4; ++nt) {
            // B-fragment: lane d = nt*16+mr, s run kb+16g..+15 contiguous in Vt
            const v16h vb = ld16h(vhead + (long)(nt*16 + mr) * TT + kb + 16*g);
            acc[nt] = __builtin_amdgcn_wmma_f32_16x16x32_f16(false, pa, false, vb,
                                                             (short)0, acc[nt],
                                                             false, false);
        }
    }
    // cross-wave reduction (ds_add_f32)
#pragma unroll
    for (int nt = 0; nt < 4; ++nt)
#pragma unroll
        for (int rr = 0; rr < 8; ++rr)
            atomicAdd(&outacc[(rr + 8*g)*64 + nt*16 + mr], acc[nt][rr]);
    __syncthreads();

    // ---- write attention output tile (f16) ----
    for (int i = tid; i < 16*64; i += 128) {
        const int m = i >> 6;
        const int d = i & 63;
        attn_out[(long)(b*TT + t0 + m) * DIMM + h*DKK + d] = (_Float16)outacc[i];
    }
}

// ----------------------------------------------------------------- launch ---
extern "C" void kernel_launch(void* const* d_in, const int* in_sizes, int n_in,
                              void* d_out, int out_size, void* d_ws, size_t ws_size,
                              hipStream_t stream)
{
    const float* x     = (const float*)d_in[0];   // [B,T,512]
    const float* Wqkv  = (const float*)d_in[1];   // [512,1536]
    const float* Wout  = (const float*)d_in[2];   // [512,512]
    const float* alpha = (const float*)d_in[3];   // [8]
    float* out = (float*)d_out;                   // [B,T,512] f32

    const int MROWS = BB * TT;                    // 8192

    // workspace carve-up (all f16)
    _Float16* Wqkv_t = (_Float16*)d_ws;                       // [1536,512]
    _Float16* Wout_t = Wqkv_t + (long)C3 * DIMM;              // [512,512]
    _Float16* xh     = Wout_t + (long)DIMM * DIMM;            // [8192,512]
    _Float16* qkv_h  = xh     + (long)MROWS * DIMM;           // [8192,1536]
    _Float16* attn_h = qkv_h  + (long)MROWS * C3;             // [8192,512]
    _Float16* vt     = attn_h + (long)MROWS * DIMM;           // [B,H,64,T]

    // 1) convert / transpose inputs to f16
    {
        int n = MROWS * DIMM;
        f32_to_f16_kernel<<<(n + 255) / 256, 256, 0, stream>>>(x, xh, n);
        n = C3 * DIMM;
        f32_to_f16_transpose_kernel<<<(n + 255) / 256, 256, 0, stream>>>(Wqkv, Wqkv_t, DIMM, C3);
        n = DIMM * DIMM;
        f32_to_f16_transpose_kernel<<<(n + 255) / 256, 256, 0, stream>>>(Wout, Wout_t, DIMM, DIMM);
    }

    // 2) qkv = xh @ Wqkv  (M=8192, N=1536, K=512) -> f16, 16x64 tiles/wave
    {
        const int tiles  = (MROWS / 16) * (C3 / 64);          // 12288 waves
        const int blocks = (tiles * 32 + 255) / 256;          // 1536
        gemm_f16_wmma<true><<<blocks, 256, 0, stream>>>(xh, Wqkv_t, qkv_h,
                                                        MROWS, C3, DIMM);
    }

    // 3) extract + transpose V for vectorized PV operand loads
    {
        const int n = BB * HH * DKK * TT;
        v_transpose_kernel<<<(n + 255) / 256, 256, 0, stream>>>(qkv_h, vt);
    }

    // 4) attention: one block per (b, h, 16-query tile); ~133 KB dynamic LDS
    {
        const int blocks = BB * HH * (TT / 16);               // 4096
        const size_t smem = (size_t)(16*2048 + 128 + 16*5 + 16*64) * sizeof(float);
        attn_topk_kernel<<<blocks, 128, smem, stream>>>(qkv_h, vt, alpha, attn_h);
    }

    // 5) out = attn_h @ Wout  (M=8192, N=512, K=512) -> f32
    {
        const int tiles  = (MROWS / 16) * (DIMM / 64);        // 4096 waves
        const int blocks = (tiles * 32 + 255) / 256;          // 512
        gemm_f16_wmma<false><<<blocks, 256, 0, stream>>>(attn_h, Wout_t, out,
                                                         MROWS, DIMM, DIMM);
    }
}
